// SAGEEdgePredictor_85744727097864
// MI455X (gfx1250) — matrix operand order repaired and checked by reading, not compiled, not run on hardware
//
#include <hip/hip_runtime.h>
#include <math.h>

#define NN  100000
#define NE  800000
#define DIM 128

typedef __attribute__((ext_vector_type(16))) __bf16 v16bf;
typedef __attribute__((ext_vector_type(8)))  float  v8f;

union Frag {
    v16bf v;
    unsigned short s[16];
    uint4 q[2];
};

__device__ __forceinline__ unsigned short f2bf(float f) {
    union { float f; unsigned u; } a; a.f = f;
    // round-to-nearest-even fp32 -> bf16
    return (unsigned short)((a.u + 0x7FFFu + ((a.u >> 16) & 1u)) >> 16);
}

// A fragment (16x32 bf16) from a row-major [16][128] bf16 tile (LDS or global).
// lane<16: K = k0+0..7, k0+16..23 ; lane>=16: K = k0+8..15, k0+24..31
__device__ __forceinline__ void load_a_tile(Frag& f, const unsigned short* tile, int k0, int lane) {
    int row = lane & 15;
    int kb  = (lane < 16) ? 0 : 8;
    const unsigned short* p = tile + row * DIM + k0 + kb;
    f.q[0] = *(const uint4*)(p);
    f.q[1] = *(const uint4*)(p + 16);
}

// A fragment straight from a gathered bf16 feature row (K-contiguous).
__device__ __forceinline__ void load_a_row(Frag& f, const unsigned short* row, int k0, int lane) {
    int kb = (lane < 16) ? 0 : 8;
    const unsigned short* p = row + k0 + kb;
    f.q[0] = *(const uint4*)(p);
    f.q[1] = *(const uint4*)(p + 16);
}

// B fragment (32x16 bf16). B[k][n] = W[n][k], W row-major [outDim][K] bf16.
// lane holds column n0+(l&15); s[0..7] = K[k0+kb+0..7], s[8..15] = K[k0+kb+16..23], kb = l<16 ? 0 : 8
__device__ __forceinline__ void load_b(Frag& f, const unsigned short* W, int K, int n0, int k0, int lane) {
    int col = n0 + (lane & 15);
    int kb  = (lane < 16) ? 0 : 8;
    const unsigned short* p = W + (size_t)col * K + k0 + kb;
    f.q[0] = *(const uint4*)(p);
    f.q[1] = *(const uint4*)(p + 16);
}

__device__ __forceinline__ v8f wmma_bf16(const Frag& a, const Frag& b, v8f c) {
    return __builtin_amdgcn_wmma_f32_16x16x32_bf16(false, a.v, false, b.v, (short)0, c, false, false);
}

// ---------------- aggregation: wave-per-edge scatter-add -----------------
__global__ void scatter_edges(const float* __restrict__ feat, const int* __restrict__ ei,
                              float* __restrict__ agg, float* __restrict__ deg) {
    int wv = threadIdx.x >> 5;
    int lane = threadIdx.x & 31;
    int e = blockIdx.x * 8 + wv;
    int s = ei[e];
    int d = ei[NE + e];
    const float* xs = feat + (size_t)s * DIM;
    float* ad = agg + (size_t)d * DIM;
#pragma unroll
    for (int j = 0; j < 4; ++j) {
        int f = lane + j * 32;
        atomicAdd(&ad[f], xs[f]);
    }
    if (deg != nullptr && lane == 0) atomicAdd(&deg[d], 1.0f);
}

__global__ void finalize_deg(float* __restrict__ deg) {
    int i = blockIdx.x * blockDim.x + threadIdx.x;
    if (i < NN) deg[i] = 1.0f / fmaxf(deg[i], 1.0f);
}

__global__ void cvt_f32_bf16(const float* __restrict__ in, unsigned short* __restrict__ out, int n) {
    int i = blockIdx.x * blockDim.x + threadIdx.x;
    if (i < n) out[i] = f2bf(in[i]);
}

// ------------- SAGE layer: out = relu(mean_agg@Wl.T + b + self@Wr.T) -----
// grid: NN/16 node tiles; 256 threads = 8 waves; A tiles converted to bf16
// once into LDS and shared by all 8 waves (wave w owns output N-tile w).
template <bool BF16OUT>
__global__ void sage_gemm(const float* __restrict__ agg, const float* __restrict__ invdeg,
                          const float* __restrict__ self,
                          const unsigned short* __restrict__ Wl, const unsigned short* __restrict__ Wr,
                          const float* __restrict__ bias,
                          float* __restrict__ outf, unsigned short* __restrict__ outb) {
    __shared__ unsigned short sA[2 * 16 * DIM];   // 8 KB: [0]=scaled agg, [1]=self
    int t    = threadIdx.x;
    int lane = t & 31;
    int wv   = t >> 5;                 // N-tile 0..7
    int tile = blockIdx.x;             // node tile (16 nodes)

    // cooperative fp32->bf16 staging: thread t handles row t>>4, cols (t&15)*8..+7
    {
        int row = t >> 4;
        int c8  = (t & 15) * 8;
        int node = (tile << 4) + row;
        float sc = invdeg[node];
        const float* p1 = agg  + (size_t)node * DIM + c8;
        const float* p2 = self + (size_t)node * DIM + c8;
        float4 u0 = ((const float4*)p1)[0], u1 = ((const float4*)p1)[1];
        float4 v0 = ((const float4*)p2)[0], v1 = ((const float4*)p2)[1];
        unsigned short* d1 = sA + row * DIM + c8;
        unsigned short* d2 = sA + 16 * DIM + row * DIM + c8;
        d1[0] = f2bf(u0.x * sc); d1[1] = f2bf(u0.y * sc);
        d1[2] = f2bf(u0.z * sc); d1[3] = f2bf(u0.w * sc);
        d1[4] = f2bf(u1.x * sc); d1[5] = f2bf(u1.y * sc);
        d1[6] = f2bf(u1.z * sc); d1[7] = f2bf(u1.w * sc);
        d2[0] = f2bf(v0.x); d2[1] = f2bf(v0.y);
        d2[2] = f2bf(v0.z); d2[3] = f2bf(v0.w);
        d2[4] = f2bf(v1.x); d2[5] = f2bf(v1.y);
        d2[6] = f2bf(v1.z); d2[7] = f2bf(v1.w);
    }
    __syncthreads();

    v8f acc = {};
    Frag a, b;
#pragma unroll
    for (int kt = 0; kt < 4; ++kt) {
        load_a_tile(a, sA, kt * 32, lane);
        load_b(b, Wl, DIM, wv * 16, kt * 32, lane);
        acc = wmma_bf16(a, b, acc);
    }
#pragma unroll
    for (int kt = 0; kt < 4; ++kt) {
        load_a_tile(a, sA + 16 * DIM, kt * 32, lane);
        load_b(b, Wr, DIM, wv * 16, kt * 32, lane);
        acc = wmma_bf16(a, b, acc);
    }
    int col  = wv * 16 + (lane & 15);
    float bv = bias[col];
    int rowD = (tile << 4) + ((lane < 16) ? 0 : 8);  // D row = r + 8*(lane>=16)
#pragma unroll
    for (int r = 0; r < 8; ++r) {
        float v = acc[r] + bv;
        v = v > 0.0f ? v : 0.0f;
        if (BF16OUT) outb[(size_t)(rowD + r) * DIM + col] = f2bf(v);
        else         outf[(size_t)(rowD + r) * DIM + col] = v;
    }
}

// ------------- edge MLP: sigmoid(relu([h_s,h_d]@Wm1.T + bm1)@Wm2.T + bm2) --
// one wave per 16 edges; 8 waves/block => 128 edges/block.
// Wm1 (64 KB bf16) async-staged into LDS once per block; kt-outer loop keeps
// 8 accumulators resident so A/B fragments are transient (no spills).
__global__ void __launch_bounds__(256)
edge_mlp(const unsigned short* __restrict__ h2, const int* __restrict__ ei,
         const unsigned short* __restrict__ Wm1b, const float* __restrict__ bm1,
         const float* __restrict__ Wm2, const float* __restrict__ bm2,
         float* __restrict__ out) {
    __shared__ unsigned short sW[DIM * 2 * DIM];  // 64 KB, row-major [128][256]

    // async global -> LDS staging of Wm1 (4096 x b128; 16 per thread)
    {
        unsigned t = threadIdx.x;
        unsigned ldsbase = (unsigned)(uintptr_t)(&sW[0]);   // low 32 bits = LDS offset
#pragma unroll
        for (int i = 0; i < 16; ++i) {
            unsigned idx  = t + (unsigned)i * 256u;         // 16-byte chunk index
            unsigned loff = ldsbase + idx * 16u;
            const unsigned short* g = Wm1b + (size_t)idx * 8;
            asm volatile("global_load_async_to_lds_b128 %0, %1, off"
                         :: "v"(loff), "v"(g) : "memory");
        }
        asm volatile("s_wait_asynccnt 0x0" ::: "memory");
    }
    __syncthreads();

    int lane = threadIdx.x & 31;
    int wv   = threadIdx.x >> 5;
    int tile = blockIdx.x * 8 + wv;       // 0..49999
    int ebase = tile * 16;
    int erow  = ebase + (lane & 15);
    int esrc  = ei[erow];
    int edst  = ei[NE + erow];
    const unsigned short* srow = h2 + (size_t)esrc * DIM;
    const unsigned short* drow = h2 + (size_t)edst * DIM;

    v8f acc[8] = {};
    Frag a, b;
#pragma unroll
    for (int kt = 0; kt < 8; ++kt) {
        // K 0..127 comes from the src feature row, 128..255 from the dst row
        load_a_row(a, (kt < 4) ? srow : drow, (kt & 3) * 32, lane);
#pragma unroll
        for (int nt = 0; nt < 8; ++nt) {
            load_b(b, sW, 2 * DIM, nt * 16, kt * 32, lane);
            acc[nt] = wmma_bf16(a, b, acc[nt]);
        }
    }

    float partial[8] = {0.f, 0.f, 0.f, 0.f, 0.f, 0.f, 0.f, 0.f};
#pragma unroll
    for (int nt = 0; nt < 8; ++nt) {
        int col  = nt * 16 + (lane & 15);
        float bv = bm1[col];
        float w2 = Wm2[col];
#pragma unroll
        for (int r = 0; r < 8; ++r) {
            float h = acc[nt][r] + bv;
            h = h > 0.0f ? h : 0.0f;
            partial[r] += h * w2;       // fused hidden @ Wm2.T
        }
    }
    // reduce over the 16 lanes sharing each D row (xor 1,2,4,8 stays in half)
#pragma unroll
    for (int m = 1; m <= 8; m <<= 1) {
#pragma unroll
        for (int r = 0; r < 8; ++r) partial[r] += __shfl_xor(partial[r], m, 32);
    }
    float b2 = bm2[0];
    if (lane == 0 || lane == 16) {
        int rb = ebase + ((lane == 0) ? 0 : 8);
#pragma unroll
        for (int r = 0; r < 8; ++r) {
            float z = partial[r] + b2;
            out[rb + r] = 1.0f / (1.0f + __expf(-z));
        }
    }
}

extern "C" void kernel_launch(void* const* d_in, const int* in_sizes, int n_in,
                              void* d_out, int out_size, void* d_ws, size_t ws_size,
                              hipStream_t stream) {
    (void)in_sizes; (void)n_in; (void)out_size; (void)ws_size;
    const float* x   = (const float*)d_in[0];
    const int*   ei  = (const int*)  d_in[1];
    const float* W1l = (const float*)d_in[2];
    const float* b1l = (const float*)d_in[3];
    const float* W1r = (const float*)d_in[4];
    const float* W2l = (const float*)d_in[5];
    const float* b2l = (const float*)d_in[6];
    const float* W2r = (const float*)d_in[7];
    const float* Wm1 = (const float*)d_in[8];
    const float* bm1 = (const float*)d_in[9];
    const float* Wm2 = (const float*)d_in[10];
    const float* bm2 = (const float*)d_in[11];
    float* out = (float*)d_out;

    char* ws = (char*)d_ws;
    size_t off = 0;
    auto alloc = [&](size_t bytes) -> void* {
        off = (off + 255) & ~(size_t)255;
        void* p = ws + off;
        off += bytes;
        return p;
    };
    float* invdeg = (float*)alloc((size_t)NN * 4);
    float* agg    = (float*)alloc((size_t)NN * DIM * 4);
    float* h1     = (float*)alloc((size_t)NN * DIM * 4);
    unsigned short* h2   = (unsigned short*)alloc((size_t)NN * DIM * 2);
    unsigned short* w1lb = (unsigned short*)alloc((size_t)DIM * DIM * 2);
    unsigned short* w1rb = (unsigned short*)alloc((size_t)DIM * DIM * 2);
    unsigned short* w2lb = (unsigned short*)alloc((size_t)DIM * DIM * 2);
    unsigned short* w2rb = (unsigned short*)alloc((size_t)DIM * DIM * 2);
    unsigned short* wm1b = (unsigned short*)alloc((size_t)DIM * 2 * DIM * 2);

    hipMemsetAsync(invdeg, 0, (size_t)NN * 4, stream);
    hipMemsetAsync(agg, 0, (size_t)NN * DIM * 4, stream);

    const int CT = 256;
    cvt_f32_bf16<<<(DIM * DIM + CT - 1) / CT, CT, 0, stream>>>(W1l, w1lb, DIM * DIM);
    cvt_f32_bf16<<<(DIM * DIM + CT - 1) / CT, CT, 0, stream>>>(W1r, w1rb, DIM * DIM);
    cvt_f32_bf16<<<(DIM * DIM + CT - 1) / CT, CT, 0, stream>>>(W2l, w2lb, DIM * DIM);
    cvt_f32_bf16<<<(DIM * DIM + CT - 1) / CT, CT, 0, stream>>>(W2r, w2rb, DIM * DIM);
    cvt_f32_bf16<<<(DIM * 2 * DIM + CT - 1) / CT, CT, 0, stream>>>(Wm1, wm1b, DIM * 2 * DIM);

    // layer 1
    scatter_edges<<<NE / 8, 256, 0, stream>>>(x, ei, agg, invdeg);
    finalize_deg<<<(NN + 255) / 256, 256, 0, stream>>>(invdeg);
    sage_gemm<false><<<NN / 16, 256, 0, stream>>>(agg, invdeg, x, w1lb, w1rb, b1l, h1, nullptr);

    // layer 2
    hipMemsetAsync(agg, 0, (size_t)NN * DIM * 4, stream);
    scatter_edges<<<NE / 8, 256, 0, stream>>>(h1, ei, agg, nullptr);
    sage_gemm<true><<<NN / 16, 256, 0, stream>>>(agg, invdeg, h1, w2lb, w2rb, b2l, nullptr, h2);

    // edge predictor
    edge_mlp<<<NE / 128, 256, 0, stream>>>(h2, ei, wm1b, bm1, Wm2, bm2, out);
}
